// Encoder_41446434407108
// MI455X (gfx1250) — matrix operand order repaired and checked by reading, not compiled, not run on hardware
//
#include <hip/hip_runtime.h>
#include <hip/hip_bf16.h>
#include <math.h>
#include <stdint.h>

typedef __attribute__((ext_vector_type(16))) __bf16 v16bf;
typedef __attribute__((ext_vector_type(8)))  float  v8f;

union V16 { v16bf v; __bf16 e[16]; };
union F8  { v8f  v; float e[8]; };

// ---------------------------------------------------------------------------
// fp32 -> bf16 conversion
// ---------------------------------------------------------------------------
__global__ void cvt_f32_bf16(const float* __restrict__ in, __bf16* __restrict__ out, int n) {
    int i = blockIdx.x * 256 + threadIdx.x;
    if (i < n) out[i] = (__bf16)in[i];
}

// ---------------------------------------------------------------------------
// Fold BN into per-channel scale/bias. Layers 0..4: scale = g/sqrt(v+eps),
// bias = b - m*scale. Layer 5 (1x1 conv): scale = 1, bias = b6.
// ---------------------------------------------------------------------------
__global__ void bnprep(const float* __restrict__ gamma, const float* __restrict__ beta,
                       const float* __restrict__ mean,  const float* __restrict__ var,
                       const float* __restrict__ b6,
                       float* __restrict__ scale, float* __restrict__ bias) {
    int t = blockIdx.x * 256 + threadIdx.x;
    if (t < 5 * 768) {
        float inv = gamma[t] * rsqrtf(var[t] + 1e-5f);
        scale[t] = inv;
        bias[t]  = beta[t] - mean[t] * inv;
    } else if (t < 5 * 768 + 64) {
        int c = t - 5 * 768;
        scale[t] = 1.0f;
        bias[t]  = b6[c];
    }
}

__global__ void zero_counts(int* __restrict__ counts) {
    int t = blockIdx.x * 256 + threadIdx.x;
    if (t < 512) counts[t] = 0;
}

// ---------------------------------------------------------------------------
// Implicit-GEMM conv1d + BN + ReLU, bf16 WMMA / fp32 accumulation.
// Fully templated so all index divides fold to mul/shift.
//
// Block: 256 threads = 8 wave32s arranged 2(M) x 4(N); tile 128(M) x 128(N).
// Each wave: 64x32 outputs = 4 M-frags x 2 N-frags = 8 WMMAs per 32-deep
// K step. LDS: A [m][32k] rows contiguous (async-staged from global for the
// aligned big layers); B transposed to [n][40k] (80B pitch, 16B aligned) so
// every fragment half is one ds_load_b128.
// ---------------------------------------------------------------------------
template<int CIN, int COUT, int TIN, int TOUT, int KW, int STRIDE, int PAD, int FINAL>
__global__ __launch_bounds__(256)
void conv_wmma(const __bf16* __restrict__ in, const __bf16* __restrict__ w,
               const float* __restrict__ scale, const float* __restrict__ bias,
               __bf16* __restrict__ out_bf, float* __restrict__ out_f32) {
    constexpr int  KT      = CIN * KW;
    constexpr int  KSTEPS  = (KT + 31) / 32;
    constexpr int  NTOT    = 32 * TOUT;
    constexpr bool USE_ASYNC = ((KT % 32) == 0) && ((COUT % 128) == 0);

    const int tid  = threadIdx.x;
    const int wave = tid >> 5;
    const int lane = tid & 31;
    const int hi   = lane >> 4;
    const int lo   = lane & 15;
    const int mw   = wave >> 2;        // 0..1  (M half)
    const int nw   = wave & 3;         // 0..3  (N quarter)
    const int m0   = blockIdx.x * 128;
    const int n0   = blockIdx.y * 128;

    __shared__ __align__(16) __bf16 sA[128 * 32];   // [m][k]   8 KB
    __shared__ __align__(16) __bf16 sB[128 * 40];   // [n][k]  10 KB (80B pitch)

    F8 acc[4][2];
    #pragma unroll
    for (int m = 0; m < 4; ++m)
        #pragma unroll
        for (int f = 0; f < 2; ++f)
            #pragma unroll
            for (int r = 0; r < 8; ++r) acc[m][f].e[r] = 0.0f;

    for (int s = 0; s < KSTEPS; ++s) {
        const int kbase = s << 5;

        // ---- stage A tile (128 rows x 32 k) ----
        if constexpr (USE_ASYNC) {
            // CDNA5 async copy: global -> LDS, no VGPR round trip (ASYNCcnt).
            // Each thread moves 32 contiguous bytes (half a 64B row).
            const int row  = tid >> 1;
            const int half = tid & 1;
            const __bf16* gp = w + (size_t)(m0 + row) * KT + kbase + half * 16;
            unsigned lds = (unsigned)(uintptr_t)(&sA[0]) + (unsigned)tid * 32u;
            unsigned long long ga = (unsigned long long)(uintptr_t)gp;
            asm volatile("global_load_async_to_lds_b128 %0, %1, off"
                         :: "v"(lds), "v"(ga) : "memory");
            asm volatile("global_load_async_to_lds_b128 %0, %1, off offset:16"
                         :: "v"(lds), "v"(ga) : "memory");
        } else {
            for (int e = tid; e < 128 * 32; e += 256) {
                int row = e >> 5, c = e & 31;
                int kt = kbase + c;
                __bf16 v = (__bf16)0.0f;
                if (kt < KT && (m0 + row) < COUT)
                    v = w[(size_t)(m0 + row) * KT + kt];
                sA[(row << 5) + c] = v;
            }
        }

        // ---- stage B tile (32 k x 128 n), transposed into [n][k] ----
        for (int e = tid; e < 32 * 128; e += 256) {
            int kk = e >> 7, nn = e & 127;          // consecutive tid -> consecutive n (coalesced)
            int kt = kbase + kk;
            int n  = n0 + nn;
            __bf16 v = (__bf16)0.0f;
            if (kt < KT && n < NTOT) {
                int ci = kt / KW, k = kt - ci * KW;         // constexpr divisors
                int b  = n / TOUT, t = n - b * TOUT;
                int tin = t * STRIDE + k - PAD;
                if (tin >= 0 && tin < TIN)
                    v = in[((size_t)b * CIN + ci) * TIN + tin];
            }
            sB[nn * 40 + kk] = v;
        }

        if constexpr (USE_ASYNC)
            asm volatile("s_wait_asynccnt 0" ::: "memory");
        __syncthreads();

        // ---- B fragments: two ds_load_b128 each (K = hi*8 + j, 16 + hi*8 + j) ----
        V16 bfr[2];
        #pragma unroll
        for (int f = 0; f < 2; ++f) {
            const __bf16* p = &sB[(nw * 32 + f * 16 + lo) * 40 + hi * 8];
            *(uint4*)&bfr[f].e[0] = *(const uint4*)p;
            *(uint4*)&bfr[f].e[8] = *(const uint4*)(p + 16);
        }
        // ---- 4 A fragments, 2 WMMAs each ----
        #pragma unroll
        for (int m = 0; m < 4; ++m) {
            V16 af;
            const __bf16* p = &sA[((mw * 64 + m * 16 + lo) << 5) + hi * 8];
            *(uint4*)&af.e[0] = *(const uint4*)p;        // K = hi*8 + (0..7)
            *(uint4*)&af.e[8] = *(const uint4*)(p + 16); // K = 16 + hi*8 + (0..7)
            acc[m][0].v = __builtin_amdgcn_wmma_f32_16x16x32_bf16(
                false, af.v, false, bfr[0].v, (short)0, acc[m][0].v, false, false);
            acc[m][1].v = __builtin_amdgcn_wmma_f32_16x16x32_bf16(
                false, af.v, false, bfr[1].v, (short)0, acc[m][1].v, false, false);
        }
        __syncthreads();
    }

    // ---- epilogue: BN scale/bias (+ReLU, bf16) or fp32 transposed z ----
    #pragma unroll
    for (int f = 0; f < 2; ++f) {
        const int n = n0 + nw * 32 + f * 16 + lo;
        if (n < NTOT) {
            const int b = n / TOUT, t = n - b * TOUT;
            #pragma unroll
            for (int m = 0; m < 4; ++m) {
                #pragma unroll
                for (int r = 0; r < 8; ++r) {
                    const int co = m0 + mw * 64 + m * 16 + hi * 8 + r;
                    if (co < COUT) {
                        float y = fmaf(acc[m][f].e[r], scale[co], bias[co]);
                        if (FINAL) {
                            out_f32[(size_t)n * 64 + co] = y;     // z[b*Tp+t, d]
                        } else {
                            y = fmaxf(y, 0.0f);
                            out_bf[((size_t)b * COUT + co) * TOUT + t] = (__bf16)y;
                        }
                    }
                }
            }
        }
    }
}

// ---------------------------------------------------------------------------
// VQ: nearest codebook entry per row of z [16352, 64], codebook [512, 64].
// Codebook staged 128 rows at a time in LDS. Deterministic reductions.
// ---------------------------------------------------------------------------
__global__ void vq_kernel(const float* __restrict__ z, const float* __restrict__ emb,
                          float* __restrict__ qout, int* __restrict__ counts,
                          float* __restrict__ blockSse) {
    const int tid = threadIdx.x;
    const int n   = blockIdx.x * 256 + tid;
    const bool valid = n < 16352;

    __shared__ float se[128 * 65];
    __shared__ float red[256];

    float x[64];
    #pragma unroll
    for (int d = 0; d < 64; ++d)
        x[d] = valid ? z[(size_t)n * 64 + d] : 0.0f;

    float best = 3.4e38f;
    int   bi   = 0;
    for (int c0 = 0; c0 < 512; c0 += 128) {
        __syncthreads();
        for (int e = tid; e < 128 * 64; e += 256) {
            int m = e >> 6, d = e & 63;
            se[m * 65 + d] = emb[(size_t)(c0 + m) * 64 + d];
        }
        __syncthreads();
        for (int m = 0; m < 128; ++m) {
            float d2 = 0.0f;
            #pragma unroll
            for (int d = 0; d < 64; ++d) {
                float df = x[d] - se[m * 65 + d];
                d2 = fmaf(df, df, d2);
            }
            if (d2 < best) { best = d2; bi = c0 + m; }
        }
    }

    if (valid) {
        #pragma unroll
        for (int d = 0; d < 64; ++d)
            qout[(size_t)n * 64 + d] = emb[(size_t)bi * 64 + d];  // q_st == quantized
        atomicAdd(&counts[bi], 1);
    }

    red[tid] = valid ? best : 0.0f;
    __syncthreads();
    for (int s = 128; s > 0; s >>= 1) {
        if (tid < s) red[tid] += red[tid + s];
        __syncthreads();
    }
    if (tid == 0) blockSse[blockIdx.x] = red[0];
}

// ---------------------------------------------------------------------------
// Loss + perplexity
// ---------------------------------------------------------------------------
__global__ void finalize(const int* __restrict__ counts, const float* __restrict__ blockSse,
                         float* __restrict__ out) {
    __shared__ float red[512];
    const int t = threadIdx.x;
    float p = (float)counts[t] / 16352.0f;
    red[t] = p * logf(p + 1e-10f);
    __syncthreads();
    for (int s = 256; s > 0; s >>= 1) {
        if (t < s) red[t] += red[t + s];
        __syncthreads();
    }
    if (t == 0) {
        float sse = 0.0f;
        for (int i = 0; i < 64; ++i) sse += blockSse[i];    // fixed order
        out[1046528] = 0.25f * sse / (16352.0f * 64.0f);    // loss
        out[1046529] = expf(-red[0]);                       // perplexity
    }
}

// ---------------------------------------------------------------------------
extern "C" void kernel_launch(void* const* d_in, const int* in_sizes, int n_in,
                              void* d_out, int out_size, void* d_ws, size_t ws_size,
                              hipStream_t stream) {
    (void)in_sizes; (void)n_in; (void)out_size; (void)ws_size;
    const float* mels  = (const float*)d_in[0];
    const float* w1    = (const float*)d_in[1];
    const float* w2    = (const float*)d_in[2];
    const float* w3    = (const float*)d_in[3];
    const float* w4    = (const float*)d_in[4];
    const float* w5    = (const float*)d_in[5];
    const float* w6    = (const float*)d_in[6];
    const float* b6    = (const float*)d_in[7];
    const float* gamma = (const float*)d_in[8];
    const float* beta  = (const float*)d_in[9];
    const float* mean  = (const float*)d_in[10];
    const float* var   = (const float*)d_in[11];
    const float* emb   = (const float*)d_in[12];
    float* out = (float*)d_out;

    char* ws = (char*)d_ws;
    size_t off = 0;
    auto alloc = [&](size_t bytes) -> void* {
        void* p = ws + off;
        off = (off + bytes + 511) & ~(size_t)511;
        return p;
    };
    const size_t ACT = 25116672;                       // 32*768*1022 (max activation)
    __bf16* actA   = (__bf16*)alloc(2ull * ACT);
    __bf16* actB   = (__bf16*)alloc(2ull * ACT);
    __bf16* melsbf = (__bf16*)alloc(2ull * 2621440);   // 32*80*1024
    __bf16* w1b    = (__bf16*)alloc(2ull * 184320);    // 768*80*3
    __bf16* w2b    = (__bf16*)alloc(2ull * 1769472);   // 768*768*3
    __bf16* w3b    = (__bf16*)alloc(2ull * 2359296);   // 768*768*4
    __bf16* w4b    = (__bf16*)alloc(2ull * 1769472);
    __bf16* w5b    = (__bf16*)alloc(2ull * 1769472);
    __bf16* w6b    = (__bf16*)alloc(2ull * 49152);     // 64*768*1
    float*  scaleA = (float*)alloc(4ull * 3904);
    float*  biasA  = (float*)alloc(4ull * 3904);
    float*  zbuf   = (float*)alloc(4ull * 1046528);    // 16352*64
    int*    counts = (int*)alloc(4ull * 512);
    float*  bSse   = (float*)alloc(4ull * 64);

    // prep: conversions, BN fold, count reset
    cvt_f32_bf16<<<(2621440 + 255) / 256, 256, 0, stream>>>(mels, melsbf, 2621440);
    cvt_f32_bf16<<<(184320  + 255) / 256, 256, 0, stream>>>(w1, w1b, 184320);
    cvt_f32_bf16<<<(1769472 + 255) / 256, 256, 0, stream>>>(w2, w2b, 1769472);
    cvt_f32_bf16<<<(2359296 + 255) / 256, 256, 0, stream>>>(w3, w3b, 2359296);
    cvt_f32_bf16<<<(1769472 + 255) / 256, 256, 0, stream>>>(w4, w4b, 1769472);
    cvt_f32_bf16<<<(1769472 + 255) / 256, 256, 0, stream>>>(w5, w5b, 1769472);
    cvt_f32_bf16<<<(49152   + 255) / 256, 256, 0, stream>>>(w6, w6b, 49152);
    bnprep<<<16, 256, 0, stream>>>(gamma, beta, mean, var, b6, scaleA, biasA);
    zero_counts<<<2, 256, 0, stream>>>(counts);

    // conv stack: tile 128(M) x 128(N); N = 32*Tout
    conv_wmma<80, 768, 1024, 1022, 3, 1, 0, 0><<<dim3(6, 256), 256, 0, stream>>>(
        melsbf, w1b, scaleA + 0,    biasA + 0,    actA, nullptr);
    conv_wmma<768, 768, 1022, 1022, 3, 1, 1, 0><<<dim3(6, 256), 256, 0, stream>>>(
        actA,   w2b, scaleA + 768,  biasA + 768,  actB, nullptr);
    conv_wmma<768, 768, 1022, 511, 4, 2, 1, 0><<<dim3(6, 128), 256, 0, stream>>>(
        actB,   w3b, scaleA + 1536, biasA + 1536, actA, nullptr);
    conv_wmma<768, 768, 511, 511, 3, 1, 1, 0><<<dim3(6, 128), 256, 0, stream>>>(
        actA,   w4b, scaleA + 2304, biasA + 2304, actB, nullptr);
    conv_wmma<768, 768, 511, 511, 3, 1, 1, 0><<<dim3(6, 128), 256, 0, stream>>>(
        actB,   w5b, scaleA + 3072, biasA + 3072, actA, nullptr);
    conv_wmma<768, 64, 511, 511, 1, 1, 0, 1><<<dim3(1, 128), 256, 0, stream>>>(
        actA,   w6b, scaleA + 3840, biasA + 3840, nullptr, zbuf);

    // vector quantization + stats
    vq_kernel<<<64, 256, 0, stream>>>(zbuf, emb, out, counts, bSse);
    finalize<<<1, 512, 0, stream>>>(counts, bSse, out);
}
